// Generator_14027363189239
// MI455X (gfx1250) — matrix operand order repaired and checked by reading, not compiled, not run on hardware
//
#include <hip/hip_runtime.h>

#define VOX 262144                    // 64*64*64 voxels
#define CHBUF ((size_t)512 * VOX)     // elements per ping-pong activation buffer

typedef __attribute__((ext_vector_type(16))) __bf16 v16bf;
typedef __attribute__((ext_vector_type(8)))  __bf16 v8bf;
typedef __attribute__((ext_vector_type(8)))  float  v8f;

union frag16 { v16bf v; v8bf h[2]; };

// ---------------------------------------------------------------------------
// f32 -> bf16 conversion (weights converted once per launch into d_ws).
// ---------------------------------------------------------------------------
__global__ __launch_bounds__(256) void cvt_f32_to_bf16(
    const float* __restrict__ src, __bf16* __restrict__ dst, int n)
{
    const int i = blockIdx.x * 256 + threadIdx.x;
    if (i < n) dst[i] = (__bf16)src[i];
}

// ---------------------------------------------------------------------------
// Fused pointwise GEMM over voxel-major bf16 activations X[v][ci]:
//   Y[v][co] = act( g[co]*(sum_ci W[co][ci]*X[v][ci]) + b[co] (+ R[v][co]) )
// FLAGS bit0 = relu, bit1 = residual (compile-time -> straight-line epilogue).
// Wave tile: 32 co x 64 vox = 2x4 WMMA 16x16x32 tiles; block = 4 waves (voxels).
// All fragment traffic is b128: A = 2x16B (weights), B = 2x16B (activations),
// epilogue = 1x16B packed store (+1x16B residual load) per 16x16 tile.
// ---------------------------------------------------------------------------
template <int FLAGS>
__global__ __launch_bounds__(128) void pw_gemm_bn_wmma(
    const __bf16* __restrict__ X, const __bf16* __restrict__ W,
    const float* __restrict__ g, const float* __restrict__ b,
    const __bf16* __restrict__ R, __bf16* __restrict__ Y,
    int Ci, int Co)
{
    const int lane = threadIdx.x & 31;
    const int wave = threadIdx.x >> 5;
    const int hlf  = lane >> 4;      // 0: lanes 0-15, 1: lanes 16-31
    const int l15  = lane & 15;
    const int voxBase = blockIdx.x * 256 + wave * 64;
    const int coBase  = blockIdx.y * 32;

    v8f acc[2][4] = {};

#pragma unroll 2
    for (int k = 0; k < Ci; k += 32) {
        // ---- A fragments: weight tiles (16 co x 32 ci) ---------------------
        // ISA 16-bit A layout: lanes 0-15 (M=l15): elems 0-7 = K k..k+7,
        // elems 8-15 = K k+16..k+23; lanes 16-31: K k+8..15 / k+24..31.
        frag16 afrag[2];
#pragma unroll
        for (int t = 0; t < 2; ++t) {
            const __bf16* wp = W + (size_t)(coBase + t * 16 + l15) * Ci + k + hlf * 8;
            afrag[t].h[0] = *(const v8bf*)(wp);       // b128
            afrag[t].h[1] = *(const v8bf*)(wp + 16);  // b128
        }
        // ---- B fragments: activation tiles (32 ci x 16 vox) ----------------
        // Lanes 0-15 (N=l15): elems 0-7 = X[v][k..k+7], elems 8-15 = X[v][k+16..k+23];
        // lanes 16-31: X[v][k+8..k+15] / X[v][k+24..k+31]. Two 8-elem runs = two b128 loads.
        frag16 bfrag[4];
#pragma unroll
        for (int s = 0; s < 4; ++s) {
            const __bf16* xp = X + (size_t)(voxBase + s * 16 + l15) * Ci + k + hlf * 8;
            bfrag[s].h[0] = *(const v8bf*)(xp);       // b128
            bfrag[s].h[1] = *(const v8bf*)(xp + 16);  // b128
            __builtin_prefetch(xp + 64, 0, 3);        // speculative: next 128B of row
        }
        // ---- 8 WMMAs: 2 A-frags reused across 4 B-frags --------------------
#pragma unroll
        for (int t = 0; t < 2; ++t)
#pragma unroll
            for (int s = 0; s < 4; ++s)
                acc[t][s] = __builtin_amdgcn_wmma_f32_16x16x32_bf16(
                    false, afrag[t].v, false, bfrag[s].v,
                    (short)0, acc[t][s], false, false);
    }

    // ---- Epilogue ----------------------------------------------------------
    // D layout: VGPR i, lanes 0-15 -> (M=i, N=l15); lanes 16-31 -> (M=i+8, N=l15).
    // Each lane owns 8 contiguous output channels -> packed b128 store.
    float gv[2][8], bv[2][8];
#pragma unroll
    for (int t = 0; t < 2; ++t)
#pragma unroll
        for (int i = 0; i < 8; ++i) {
            const int co = coBase + t * 16 + hlf * 8 + i;   // contiguous -> b128
            gv[t][i] = g[co];
            bv[t][i] = b[co];
        }

#pragma unroll
    for (int t = 0; t < 2; ++t) {
#pragma unroll
        for (int s = 0; s < 4; ++s) {
            const int vcol = voxBase + s * 16 + l15;
            const int co0  = coBase + t * 16 + hlf * 8;
            v8bf outv;
            v8bf rv{};
            if constexpr (FLAGS & 2)
                rv = *(const v8bf*)(R + (size_t)vcol * Co + co0);  // b128
#pragma unroll
            for (int i = 0; i < 8; ++i) {
                float val = fmaf(gv[t][i], acc[t][s][i], bv[t][i]);
                if constexpr (FLAGS & 2) val += (float)rv[i];
                if constexpr (FLAGS & 1) val = fmaxf(val, 0.0f);
                outv[i] = (__bf16)val;
            }
            *(v8bf*)(Y + (size_t)vcol * Co + co0) = outv;          // b128 store
        }
    }
}

// ---------------------------------------------------------------------------
// Stem: concat(x, broadcast(c)) -> pad 1 -> 3x3x3 VALID conv -> BN -> ReLU.
// K = 135, ~4.5 GFLOP => VALU kernel. Threads: co fastest (64/voxel) so the
// voxel-major [v*64+co] bf16 stores are fully coalesced.
// ---------------------------------------------------------------------------
__global__ __launch_bounds__(256) void stem_conv3d(
    const float* __restrict__ x, const float* __restrict__ c,
    const float* __restrict__ w, const float* __restrict__ g,
    const float* __restrict__ b, __bf16* __restrict__ out)
{
    const int co = threadIdx.x & 63;
    const int v  = blockIdx.x * 4 + (threadIdx.x >> 6);   // 4 voxels per block
    const int z  = v >> 12;
    const int y  = (v >> 6) & 63;
    const int xw = v & 63;

    float cv[4];
#pragma unroll
    for (int i = 0; i < 4; ++i) cv[i] = c[i];

    const float* wc = w + (size_t)co * 5 * 27;
    float sum = 0.0f;
#pragma unroll
    for (int kz = 0; kz < 3; ++kz) {
#pragma unroll
        for (int ky = 0; ky < 3; ++ky) {
#pragma unroll
            for (int kx = 0; kx < 3; ++kx) {
                const int zz = z + kz - 1, yy = y + ky - 1, xx = xw + kx - 1;
                const bool in = ((unsigned)zz < 64u) & ((unsigned)yy < 64u) & ((unsigned)xx < 64u);
                const int tap = kz * 9 + ky * 3 + kx;
                const float xv = in ? x[((size_t)zz * 64 + yy) * 64 + xx] : 0.0f;
                sum += wc[tap] * xv;                       // channel 0 (x volume)
#pragma unroll
                for (int ci = 1; ci < 5; ++ci)             // channels 1..4 (c broadcast)
                    sum += wc[ci * 27 + tap] * (in ? cv[ci - 1] : 0.0f);
            }
        }
    }
    const float r = g[co] * sum + b[co];
    out[(size_t)v * 64 + co] = (__bf16)fmaxf(r, 0.0f);
}

// ---------------------------------------------------------------------------
// Final FC: Co = 1 -> per-voxel dot over 512 contiguous bf16 channels (b128
// vector loads), f32 output.
// ---------------------------------------------------------------------------
__global__ __launch_bounds__(256) void fc_head(
    const __bf16* __restrict__ h, const float* __restrict__ w,
    const float* __restrict__ bias, float* __restrict__ out)
{
    const int v = blockIdx.x * 256 + threadIdx.x;
    const __bf16* hp = h + (size_t)v * 512;
    float s = bias[0];
#pragma unroll 4
    for (int ci = 0; ci < 512; ci += 8) {
        v8bf hv = *(const v8bf*)(hp + ci);                 // b128
#pragma unroll
        for (int j = 0; j < 8; ++j)
            s += (float)hv[j] * w[ci + j];
    }
    out[v] = s;
}

// ---------------------------------------------------------------------------
extern "C" void kernel_launch(void* const* d_in, const int* in_sizes, int n_in,
                              void* d_out, int out_size, void* d_ws, size_t ws_size,
                              hipStream_t stream)
{
    const float* x       = (const float*)d_in[0];
    const float* c       = (const float*)d_in[1];
    const float* w_conv1 = (const float*)d_in[2];
    const float* g1      = (const float*)d_in[3];
    const float* b1      = (const float*)d_in[4];
    const float* w1a = (const float*)d_in[5];  const float* g1a = (const float*)d_in[6];  const float* b1a = (const float*)d_in[7];
    const float* w1b = (const float*)d_in[8];  const float* g1b = (const float*)d_in[9];  const float* b1b = (const float*)d_in[10];
    const float* w2a = (const float*)d_in[11]; const float* g2a = (const float*)d_in[12]; const float* b2a = (const float*)d_in[13];
    const float* w2b = (const float*)d_in[14]; const float* g2b = (const float*)d_in[15]; const float* b2b = (const float*)d_in[16];
    const float* w2d = (const float*)d_in[17]; const float* g2d = (const float*)d_in[18]; const float* b2d = (const float*)d_in[19];
    const float* w3a = (const float*)d_in[20]; const float* g3a = (const float*)d_in[21]; const float* b3a = (const float*)d_in[22];
    const float* w3b = (const float*)d_in[23]; const float* g3b = (const float*)d_in[24]; const float* b3b = (const float*)d_in[25];
    const float* w3d = (const float*)d_in[26]; const float* g3d = (const float*)d_in[27]; const float* b3d = (const float*)d_in[28];
    const float* w4a = (const float*)d_in[29]; const float* g4a = (const float*)d_in[30]; const float* b4a = (const float*)d_in[31];
    const float* w4b = (const float*)d_in[32]; const float* g4b = (const float*)d_in[33]; const float* b4b = (const float*)d_in[34];
    const float* w4d = (const float*)d_in[35]; const float* g4d = (const float*)d_in[36]; const float* b4d = (const float*)d_in[37];
    const float* w_fc = (const float*)d_in[38];
    const float* b_fc = (const float*)d_in[39];

    // ---- workspace layout (all bf16): 3 activation buffers + weight arena --
    __bf16* buf0 = (__bf16*)d_ws;          // idn / residual path
    __bf16* buf1 = buf0 + CHBUF;           // y (inner activation)
    __bf16* buf2 = buf1 + CHBUF;           // h (block output)
    __bf16* warena = buf2 + CHBUF;

    struct { const float* src; int n; } wm[11] = {
        {w1a,  64 * 64}, {w1b,  64 * 64},
        {w2a, 128 * 64}, {w2b, 128 * 128}, {w2d, 128 * 64},
        {w3a, 256 * 128}, {w3b, 256 * 256}, {w3d, 256 * 128},
        {w4a, 512 * 256}, {w4b, 512 * 512}, {w4d, 512 * 256},
    };
    __bf16* wb[11];
    {
        __bf16* p = warena;
        for (int i = 0; i < 11; ++i) {
            wb[i] = p;
            p += wm[i].n;
            cvt_f32_to_bf16<<<dim3((wm[i].n + 255) / 256), dim3(256), 0, stream>>>(
                wm[i].src, wb[i], wm[i].n);
        }
    }

    // Stem -> buf0 (h0, 64 ch, voxel-major bf16)
    stem_conv3d<<<dim3(VOX / 4), dim3(256), 0, stream>>>(x, c, w_conv1, g1, b1, buf0);

    auto gemm = [&](const __bf16* X, const __bf16* W, const float* gg, const float* bb,
                    const __bf16* R, __bf16* Y, int Ci, int Co, int flags) {
        const dim3 grid(VOX / 256, Co / 32);
        const dim3 blk(128);
        if (flags == 3)
            pw_gemm_bn_wmma<3><<<grid, blk, 0, stream>>>(X, W, gg, bb, R, Y, Ci, Co);
        else if (flags == 1)
            pw_gemm_bn_wmma<1><<<grid, blk, 0, stream>>>(X, W, gg, bb, R, Y, Ci, Co);
        else
            pw_gemm_bn_wmma<0><<<grid, blk, 0, stream>>>(X, W, gg, bb, R, Y, Ci, Co);
    };

    // Block 1 (64 -> 64, identity residual)
    gemm(buf0, wb[0], g1a, b1a, nullptr, buf1, 64, 64, 1);
    gemm(buf1, wb[1], g1b, b1b, buf0,    buf2, 64, 64, 3);     // h1 -> buf2

    // Block 2 (64 -> 128, projection residual)
    gemm(buf2, wb[4], g2d, b2d, nullptr, buf0, 64, 128, 0);    // idn
    gemm(buf2, wb[2], g2a, b2a, nullptr, buf1, 64, 128, 1);
    gemm(buf1, wb[3], g2b, b2b, buf0,    buf2, 128, 128, 3);   // h2 -> buf2

    // Block 3 (128 -> 256)
    gemm(buf2, wb[7], g3d, b3d, nullptr, buf0, 128, 256, 0);
    gemm(buf2, wb[5], g3a, b3a, nullptr, buf1, 128, 256, 1);
    gemm(buf1, wb[6], g3b, b3b, buf0,    buf2, 256, 256, 3);   // h3 -> buf2

    // Block 4 (256 -> 512)
    gemm(buf2, wb[10], g4d, b4d, nullptr, buf0, 256, 512, 0);
    gemm(buf2, wb[8],  g4a, b4a, nullptr, buf1, 256, 512, 1);
    gemm(buf1, wb[9],  g4b, b4b, buf0,    buf2, 512, 512, 3);  // h4 -> buf2

    // FC head -> d_out (f32, 262144 elements)
    fc_head<<<dim3(VOX / 256), dim3(256), 0, stream>>>(buf2, w_fc, b_fc, (float*)d_out);
}
